// BDPool_4698694221941
// MI455X (gfx1250) — matrix operand order repaired and checked by reading, not compiled, not run on hardware
//
#include <hip/hip_runtime.h>
#include <hip/hip_bf16.h>
#include <stdint.h>

typedef __bf16 bf16;
typedef __attribute__((ext_vector_type(16))) __bf16 v16bf;
typedef __attribute__((ext_vector_type(8)))  float  v8f;

#define HW   16384   // 128*128
#define NB   8
#define ASTR 40      // LDS row stride (32 k + 8 pad) in bf16
#define BTILE (128 * ASTR)

union U16 { uint4 q[2]; bf16 h[16]; };

// Register staging for one K-tile: A chunk (8 VGPRs) + up to 3 B tap-tiles (24)
struct Stage {
    U16  a;
    uint4 b[3][2];
};

// ---------------------------------------------------------------------------
// Fused implicit-GEMM conv (3x3, pad=1) + optional extra 1x1 path, BN-folded
// weights, shift + ReLU epilogue. Software-pipelined: tile it+1 is loaded from
// global into registers while tile it computes from LDS.
// grid = (NB*128 rows, Cout/128), block = 256 (8 waves)
// OMODE: 0 = split bf16 (och<128 -> out0, else out1), 1 = bf16, 2 = f32
// ---------------------------------------------------------------------------
template<int CIN, bool EXTRA, int OMODE>
__global__ __launch_bounds__(256)
void conv3x3_wmma(const bf16* __restrict__ src, const bf16* __restrict__ wt,
                  const float* __restrict__ shift,
                  const bf16* __restrict__ src2, const bf16* __restrict__ wt2,
                  const float* __restrict__ shift2,
                  void* __restrict__ out0, void* __restrict__ out1)
{
    __shared__ bf16 a_lds[130 * ASTR];
    __shared__ bf16 b_lds[3 * BTILE];

    constexpr int NC    = CIN / 32;       // K-steps per conv row
    constexpr int TMAIN = 3 * NC;         // 3 conv rows
    constexpr int T     = TMAIN + (EXTRA ? 8 : 0);

    const int row  = blockIdx.x;
    const int nImg = row >> 7;
    const int h    = row & 127;
    const int o0   = blockIdx.y << 7;

    const int lane = threadIdx.x & 31;
    const int wv   = threadIdx.x >> 5;
    const int mb0  = (wv & 3) * 32;   // 2 M-blocks: mb0, mb0+16
    const int nb0  = (wv >> 2) * 64;  // 4 N-blocks: nb0 .. nb0+48
    const int llo  = lane & 15;
    const int lhi  = lane >> 4;

    // loader thread mapping
    const int am  = threadIdx.x & 127;   // A: pixel (w)
    const int akh = threadIdx.x >> 7;    // A: channel half (16 ch each)
    const int bo  = threadIdx.x >> 1;    // B: out-channel row
    const int bh  = threadIdx.x & 1;     // B: k half (16 each)

    // zero the two permanent padding rows (wsrc = -1 and wsrc = 128)
    if (threadIdx.x < 8) {
        const uint4 z = {0u, 0u, 0u, 0u};
        const int zr = (threadIdx.x < 4) ? 0 : 129;
        ((uint4*)(a_lds + zr * ASTR))[threadIdx.x & 3] = z;
    }

    // Load tile `it` from global memory into registers (all control uniform).
    auto load_stage = [&](int it, Stage& st) {
        const bf16* aSrc; const bf16* wtb; int cin, c0, hs, ntap; bool hok;
        if (!EXTRA || it < TMAIN) {
            const int r = it / NC, ci = it % NC;
            c0  = ci * 32; cin = CIN;
            hs  = h + r - 1; hok = (unsigned)hs < 128u;
            aSrc = src; wtb = wt + (size_t)r * 3 * 256 * CIN; ntap = 3;
        } else {
            c0  = (it - TMAIN) * 32; cin = 256;
            hs  = h; hok = true;
            aSrc = src2; wtb = wt2; ntap = 1;
        }
        if (hok) {
            const bf16* base = aSrc + (((size_t)nImg * cin + c0 + akh * 16) * 128 + hs) * 128 + am;
            if (c0 + 32 < cin) __builtin_prefetch(base + (size_t)32 * HW, 0, 1);
#pragma unroll
            for (int j = 0; j < 16; ++j) st.a.h[j] = base[(size_t)j * HW];
        } else {
#pragma unroll
            for (int j = 0; j < 16; ++j) st.a.h[j] = (bf16)0.0f;
        }
#pragma unroll
        for (int ss = 0; ss < 3; ++ss) {
            if (ss < ntap) {
                const uint4* g = (const uint4*)(wtb + (size_t)ss * 256 * cin +
                                                (size_t)(o0 + bo) * cin + c0 + bh * 16);
                st.b[ss][0] = g[0];
                st.b[ss][1] = g[1];
            }
        }
    };

    // Publish registers -> LDS (between the two barriers of an iteration).
    auto store_stage = [&](int it, const Stage& st) {
        uint4* ad = (uint4*)(a_lds + (am + 1) * ASTR + akh * 16);
        ad[0] = st.a.q[0];
        ad[1] = st.a.q[1];
        const int ntap = (!EXTRA || it < TMAIN) ? 3 : 1;
#pragma unroll
        for (int ss = 0; ss < 3; ++ss) {
            if (ss < ntap) {
                uint4* bd = (uint4*)(b_lds + ss * BTILE + bo * ASTR + bh * 16);
                bd[0] = st.b[ss][0];
                bd[1] = st.b[ss][1];
            }
        }
    };

    v8f acc[2][4];
    const v8f vzero = {0.f, 0.f, 0.f, 0.f, 0.f, 0.f, 0.f, 0.f};
#pragma unroll
    for (int i = 0; i < 2; ++i)
#pragma unroll
        for (int j = 0; j < 4; ++j) acc[i][j] = vzero;

    // one 16x16x32 sub-GEMM sweep; sOff selects the w-shifted A rows (m + sOff)
    auto mma_step = [&](int sOff, const bf16* __restrict__ bbase) {
        v16bf af[2], bfm[4];
#pragma unroll
        for (int i = 0; i < 2; ++i) {
            U16 ua;
            const uint4* ap = (const uint4*)(a_lds + (mb0 + i * 16 + llo + sOff) * ASTR + lhi * 8);
            ua.q[0] = ap[0];   // K = koff .. koff+7
            ua.q[1] = ap[2];   // K = koff+16 .. koff+23
            af[i] = *(v16bf*)&ua;
        }
#pragma unroll
        for (int j = 0; j < 4; ++j) {
            U16 ub;
            const uint4* bp = (const uint4*)(bbase + (nb0 + j * 16 + llo) * ASTR + lhi * 8);
            ub.q[0] = bp[0];
            ub.q[1] = bp[2];
            bfm[j] = *(v16bf*)&ub;
        }
#pragma unroll
        for (int i = 0; i < 2; ++i)
#pragma unroll
            for (int j = 0; j < 4; ++j)
                acc[i][j] = __builtin_amdgcn_wmma_f32_16x16x32_bf16(
                    false, af[i], false, bfm[j], (short)0, acc[i][j], false, false);
    };

    auto compute_tile = [&](int it) {
        if (!EXTRA || it < TMAIN) {
            mma_step(0, b_lds);
            mma_step(1, b_lds + BTILE);
            mma_step(2, b_lds + 2 * BTILE);
        } else {
            mma_step(1, b_lds);   // 1x1 path: unshifted rows (wsrc = m)
        }
    };

    // ---- software pipeline (T is even for all instantiations) ----
    Stage s0, s1;
    load_stage(0, s0);
#pragma unroll 1
    for (int it = 0; it < T; it += 2) {
        __syncthreads();                       // prev compute done; LDS free
        store_stage(it, s0);
        if (it + 1 < T) load_stage(it + 1, s1);
        __syncthreads();                       // tile `it` visible to all
        compute_tile(it);
        if (it + 1 >= T) break;
        __syncthreads();
        store_stage(it + 1, s1);
        if (it + 2 < T) load_stage(it + 2, s0);
        __syncthreads();
        compute_tile(it + 1);
    }

    // Epilogue: D layout -> lane holds (M = g + 8*lhi, N = llo) per VGPR g
#pragma unroll
    for (int i = 0; i < 2; ++i)
#pragma unroll
        for (int j = 0; j < 4; ++j) {
            const int och = o0 + nb0 + j * 16 + llo;
            float sh = shift[och];
            if constexpr (EXTRA) sh += shift2[och];
#pragma unroll
            for (int g = 0; g < 8; ++g) {
                const int wpix = mb0 + i * 16 + lhi * 8 + g;
                float v = fmaxf(acc[i][j][g] + sh, 0.0f);
                if constexpr (OMODE == 0) {
                    if (och < 128)
                        ((bf16*)out0)[(((size_t)nImg * 128 + och) * 128 + h) * 128 + wpix] = (bf16)v;
                    else
                        ((bf16*)out1)[(((size_t)nImg * 128 + (och - 128)) * 128 + h) * 128 + wpix] = (bf16)v;
                } else if constexpr (OMODE == 1) {
                    ((bf16*)out0)[(((size_t)nImg * 256 + och) * 128 + h) * 128 + wpix] = (bf16)v;
                } else {
                    ((float*)out0)[(((size_t)nImg * 256 + och) * 128 + h) * 128 + wpix] = v;
                }
            }
        }
}

// ---------------------------------------------------------------------------
// Helpers: convert, BN-fold, corner pools
// ---------------------------------------------------------------------------
__global__ void convert_x_kernel(const float* __restrict__ x, bf16* __restrict__ xb, size_t n)
{
    size_t i = ((size_t)blockIdx.x * 256 + threadIdx.x) * 8;
    if (i >= n) return;
    float4 f0 = *(const float4*)(x + i);
    float4 f1 = *(const float4*)(x + i + 4);
    union { uint4 q; bf16 h[8]; } t;
    t.h[0] = (bf16)f0.x; t.h[1] = (bf16)f0.y; t.h[2] = (bf16)f0.z; t.h[3] = (bf16)f0.w;
    t.h[4] = (bf16)f1.x; t.h[5] = (bf16)f1.y; t.h[6] = (bf16)f1.z; t.h[7] = (bf16)f1.w;
    *(uint4*)(xb + i) = t.q;
}

// fold BN into 3x3 weights; output layout [tap][Ototal][cin] bf16 (+ shift f32)
__global__ void fold_w3_kernel(const float* __restrict__ W, const float* __restrict__ g,
                               const float* __restrict__ b, const float* __restrict__ m,
                               const float* __restrict__ v,
                               int O, int cin, int obase, int Ototal,
                               bf16* __restrict__ wt, float* __restrict__ shift)
{
    int t = blockIdx.x * 256 + threadIdx.x;
    if (t >= O * cin) return;
    int o = t / cin, c = t % cin;
    float scale = g[o] * rsqrtf(v[o] + 1e-5f);
    if (c == 0) shift[obase + o] = b[o] - m[o] * scale;
    const float* wp = W + (size_t)(o * cin + c) * 9;
#pragma unroll
    for (int tap = 0; tap < 9; ++tap)
        wt[((size_t)tap * Ototal + obase + o) * cin + c] = (bf16)(wp[tap] * scale);
}

__global__ void fold_w1_kernel(const float* __restrict__ W, const float* __restrict__ g,
                               const float* __restrict__ b, const float* __restrict__ m,
                               const float* __restrict__ v,
                               bf16* __restrict__ wt, float* __restrict__ shift)
{
    int t = blockIdx.x * 256 + threadIdx.x;   // 256*256
    int o = t >> 8, c = t & 255;
    float scale = g[o] * rsqrtf(v[o] + 1e-5f);
    if (c == 0) shift[o] = b[o] - m[o] * scale;
    wt[(size_t)o * 256 + c] = (bf16)(W[(size_t)o * 256 + c] * scale);
}

// TopPool: reverse cummax along H, in place on p1. One thread per (n,c,w).
__global__ void pool_h_kernel(bf16* __restrict__ p1)
{
    int t = blockIdx.x * 256 + threadIdx.x;      // 8*128*128
    int w = t & 127;
    int c = (t >> 7) & 127;
    int n = t >> 14;
    bf16* col = p1 + (((size_t)n * 128 + c) * 128) * 128 + w;
    float run = -3.402823466e38f;
#pragma unroll 4
    for (int h = 127; h >= 0; --h) {
        run = fmaxf(run, (float)col[(size_t)h * 128]);
        col[(size_t)h * 128] = (bf16)run;
    }
}

// LeftPool on p2 (reverse cummax along W) fused with s = pool1 + pool2 -> p1.
// One thread per (n,c,h) row.
__global__ void pool_w_add_kernel(const bf16* __restrict__ p2, bf16* __restrict__ p1)
{
    int t = blockIdx.x * 256 + threadIdx.x;      // 8*128*128 rows
    size_t base = (size_t)t * 128;
    float run = -3.402823466e38f;
#pragma unroll 1
    for (int wb = 120; wb >= 0; wb -= 8) {
        union { uint4 q; bf16 h[8]; } u2, u1;
        u2.q = *(const uint4*)(p2 + base + wb);
        u1.q = *(const uint4*)(p1 + base + wb);
#pragma unroll
        for (int k = 7; k >= 0; --k) {
            run = fmaxf(run, (float)u2.h[k]);
            u1.h[k] = (bf16)((float)u1.h[k] + run);
        }
        *(uint4*)(p1 + base + wb) = u1.q;
    }
}

// ---------------------------------------------------------------------------
extern "C" void kernel_launch(void* const* d_in, const int* in_sizes, int n_in,
                              void* d_out, int out_size, void* d_ws, size_t ws_size,
                              hipStream_t stream)
{
    (void)in_sizes; (void)n_in; (void)out_size; (void)ws_size;

    const float* x   = (const float*)d_in[0];
    const float* Wp1 = (const float*)d_in[1];
    const float *g1 = (const float*)d_in[2], *b1 = (const float*)d_in[3],
                *m1 = (const float*)d_in[4], *v1 = (const float*)d_in[5];
    const float* Wp2 = (const float*)d_in[6];
    const float *g2 = (const float*)d_in[7], *b2 = (const float*)d_in[8],
                *m2 = (const float*)d_in[9], *v2 = (const float*)d_in[10];
    const float* Wp  = (const float*)d_in[11];
    const float *gp = (const float*)d_in[12], *bp = (const float*)d_in[13],
                *mp = (const float*)d_in[14], *vp = (const float*)d_in[15];
    const float* Wc1 = (const float*)d_in[16];
    const float *gc1 = (const float*)d_in[17], *bc1 = (const float*)d_in[18],
                *mc1 = (const float*)d_in[19], *vc1 = (const float*)d_in[20];
    const float* Wc2 = (const float*)d_in[21];
    const float *gc2 = (const float*)d_in[22], *bc2 = (const float*)d_in[23],
                *mc2 = (const float*)d_in[24], *vc2 = (const float*)d_in[25];

    char* ws = (char*)d_ws;
    size_t cur = 0;
    auto wsalloc = [&](size_t bytes) -> char* {
        char* p = ws + cur; cur += (bytes + 255) & ~(size_t)255; return p;
    };
    bf16* xb    = (bf16*)wsalloc((size_t)NB * 256 * HW * 2);   // x in bf16
    bf16* p1    = (bf16*)wsalloc((size_t)NB * 128 * HW * 2);   // p1 -> pool1 -> s
    bf16* p2    = (bf16*)wsalloc((size_t)NB * 128 * HW * 2);   // p2 -> pool2
    bf16* relu  = (bf16*)wsalloc((size_t)NB * 256 * HW * 2);   // relu(p_conv+conv1)
    bf16* wtA   = (bf16*)wsalloc((size_t)9 * 256 * 256 * 2);   // W_p1 || W_p2 folded
    bf16* wtP   = (bf16*)wsalloc((size_t)9 * 256 * 128 * 2);   // W_p folded
    bf16* wtC1  = (bf16*)wsalloc((size_t)256 * 256 * 2);       // W_c1 folded
    bf16* wtD   = (bf16*)wsalloc((size_t)9 * 256 * 256 * 2);   // W_c2 folded
    float* shA  = (float*)wsalloc(256 * 4);
    float* shP  = (float*)wsalloc(256 * 4);
    float* shC1 = (float*)wsalloc(256 * 4);
    float* shD  = (float*)wsalloc(256 * 4);

    const size_t nX = (size_t)NB * 256 * HW;
    convert_x_kernel<<<(nX / 8 + 255) / 256, 256, 0, stream>>>(x, xb, nX);

    fold_w3_kernel<<<(128 * 256 + 255) / 256, 256, 0, stream>>>(Wp1, g1, b1, m1, v1, 128, 256, 0,   256, wtA, shA);
    fold_w3_kernel<<<(128 * 256 + 255) / 256, 256, 0, stream>>>(Wp2, g2, b2, m2, v2, 128, 256, 128, 256, wtA, shA);
    fold_w3_kernel<<<(256 * 128 + 255) / 256, 256, 0, stream>>>(Wp,  gp, bp, mp, vp, 256, 128, 0,   256, wtP, shP);
    fold_w1_kernel<<<(256 * 256 + 255) / 256, 256, 0, stream>>>(Wc1, gc1, bc1, mc1, vc1, wtC1, shC1);
    fold_w3_kernel<<<(256 * 256 + 255) / 256, 256, 0, stream>>>(Wc2, gc2, bc2, mc2, vc2, 256, 256, 0, 256, wtD, shD);

    dim3 cgrid(NB * 128, 2);
    // p1 || p2 = relu(bn(conv3x3(x)))  (256 concat output channels, split store)
    conv3x3_wmma<256, false, 0><<<cgrid, 256, 0, stream>>>(
        xb, wtA, shA, nullptr, nullptr, nullptr, p1, p2);

    pool_h_kernel<<<(NB * 128 * 128) / 256, 256, 0, stream>>>(p1);
    pool_w_add_kernel<<<(NB * 128 * 128) / 256, 256, 0, stream>>>(p2, p1);  // p1 := pool1+pool2

    // relu( bn(conv3x3(s)) + bn(conv1x1(x)) ) — both GEMMs share accumulators
    conv3x3_wmma<128, true, 1><<<cgrid, 256, 0, stream>>>(
        p1, wtP, shP, xb, wtC1, shC1, relu, nullptr);

    // out = relu(bn(conv3x3(relu))) in f32
    conv3x3_wmma<256, false, 2><<<cgrid, 256, 0, stream>>>(
        relu, wtD, shD, nullptr, nullptr, nullptr, d_out, nullptr);
}
